// CellEncoder_81157702025558
// MI455X (gfx1250) — matrix operand order repaired and checked by reading, not compiled, not run on hardware
//
#include <hip/hip_runtime.h>
#include <hip/hip_bf16.h>

typedef __attribute__((ext_vector_type(16))) __bf16 v16bf;
typedef __attribute__((ext_vector_type(8)))  __bf16 v8bf;
typedef __attribute__((ext_vector_type(8)))  float  v8f;
typedef __attribute__((ext_vector_type(4)))  float  v4f;

#define EMB 128   // embedding width (== in_sizes[3])

// ---------------------------------------------------------------------------
// Transpose x (f32 [Kin][N]) -> xb (bf16 [N][Kpad]), zero-padding K tail.
// LDS-tiled so both global read and global write are coalesced.
// ---------------------------------------------------------------------------
__global__ void k_transpose_bf16(const float* __restrict__ x, __bf16* __restrict__ xb,
                                 int Kin, int N, int Kpad) {
  __shared__ float tile[32][33];
  const int tx = threadIdx.x, ty = threadIdx.y;
  const int k0 = blockIdx.y * 32, c0 = blockIdx.x * 32;
#pragma unroll
  for (int i = 0; i < 32; i += 8) {
    int k = k0 + ty + i, c = c0 + tx;
    tile[ty + i][tx] = (k < Kin && c < N) ? x[(size_t)k * N + c] : 0.0f;
  }
  __syncthreads();
#pragma unroll
  for (int i = 0; i < 32; i += 8) {
    int c = c0 + ty + i, k = k0 + tx;
    if (c < N && k < Kpad) xb[(size_t)c * Kpad + k] = (__bf16)tile[tx][ty + i];
  }
}

// Weights: f32 [M][Kin] -> bf16 [M][Kpad] (zero pad K tail)
__global__ void k_wconvert(const float* __restrict__ w, __bf16* __restrict__ wb,
                           int M, int Kin, int Kpad) {
  int idx = blockIdx.x * blockDim.x + threadIdx.x;
  if (idx >= M * Kpad) return;
  int m = idx / Kpad, k = idx % Kpad;
  wb[idx] = (k < Kin) ? (__bf16)w[(size_t)m * Kin + k] : (__bf16)0.0f;
}

// ---------------------------------------------------------------------------
// WMMA GEMM: C[cell][m] = sum_k A[m][k] * B[cell][k]
//   A: bf16 [EMB][Kpad] (weights, row-major, K-contig)
//   B: bf16 [N][Kpad]   (features, K-contig per cell)
//   C: f32  [N][EMB]
// Block = 256 threads = 8 waves; wave w handles M-tile [16w,16w+16) of one
// 16-cell tile. K loop runs v_wmma_f32_16x16x32_bf16 per 32-K step.
// ---------------------------------------------------------------------------
__global__ void k_gemm_wmma(const __bf16* __restrict__ A, const __bf16* __restrict__ B,
                            float* __restrict__ C, int n_cells, int Kpad) {
  const int lane  = threadIdx.x & 31;
  const int wave  = threadIdx.x >> 5;        // 0..7 -> M tile
  const int cell0 = blockIdx.x * 16;
  const int m0    = wave * 16;
  const int r     = lane & 15;
  const int hi    = lane >> 4;               // 0 or 1 (upper lane half)

  // A lane layout: row r; halves 0-7 = K hi*8..hi*8+7, halves 8-15 = +16
  const __bf16* aptr = A + (size_t)(m0 + r) * Kpad + hi * 8;
  // B lane layout: col r; halves 0-7 = K hi*8..hi*8+7, halves 8-15 = +16
  int bcell = cell0 + r;
  if (bcell >= n_cells) bcell = n_cells - 1;     // clamp (tail tile)
  const __bf16* bptr = B + (size_t)bcell * Kpad + hi * 8;

  v8f acc = {};
#pragma unroll 4
  for (int k = 0; k < Kpad; k += 32) {
    v8bf  alo = *(const v8bf*)(aptr + k);
    v8bf  ahi = *(const v8bf*)(aptr + k + 16);
    v16bf a = __builtin_shufflevector(alo, ahi, 0,1,2,3,4,5,6,7,8,9,10,11,12,13,14,15);
    v16bf b = __builtin_shufflevector(*(const v8bf*)(bptr + k), *(const v8bf*)(bptr + k + 16), 0,1,2,3,4,5,6,7,8,9,10,11,12,13,14,15);
    acc = __builtin_amdgcn_wmma_f32_16x16x32_bf16(false, a, false, b,
                                                  (short)0, acc, false, false);
  }

  // C/D layout: VGPR j -> M = m0 + hi*8 + j (contiguous!), N = lane&15
  int ocell = cell0 + (lane & 15);
  if (ocell < n_cells) {
    float* cptr = C + (size_t)ocell * EMB + m0 + hi * 8;
    *(v8f*)cptr = acc;   // 32B aligned: two global_store_b128
  }
}

__global__ void k_zero(float* __restrict__ p, size_t n) {
  size_t i = (size_t)blockIdx.x * blockDim.x + threadIdx.x;
  if (i < n) p[i] = 0.0f;
}

__global__ void k_count(const int* __restrict__ dst, float* __restrict__ cnt, int n_edges) {
  int e = blockIdx.x * blockDim.x + threadIdx.x;
  if (e < n_edges) atomicAdd(&cnt[dst[e]], 1.0f);
}

// Scatter-add projected features: agg[dst] += p[src].  32 lanes / edge,
// 4 contiguous floats per lane (b128 gather + 4 global_atomic_add_f32).
__global__ void k_scatter(const int* __restrict__ src, const int* __restrict__ dst,
                          const float* __restrict__ p, float* __restrict__ agg,
                          int n_edges) {
  long long t = (long long)blockIdx.x * blockDim.x + threadIdx.x;
  int e = (int)(t >> 5);
  if (e >= n_edges) return;
  int l = (int)(t & 31);
  int s = src[e], d = dst[e];
  v4f v = *(const v4f*)(p + (size_t)s * EMB + l * 4);
  float* o = agg + (size_t)d * EMB + l * 4;
  atomicAdd(o + 0, v[0]);
  atomicAdd(o + 1, v[1]);
  atomicAdd(o + 2, v[2]);
  atomicAdd(o + 3, v[3]);
}

// Fused epilogue: out = elu(agg/max(cnt,1) + bias + q); optionally emit bf16
// copy (layer-0 output feeds layer-1 WMMA as B operand).
__global__ void k_finalize(const float* __restrict__ agg, const float* __restrict__ cnt,
                           const float* __restrict__ bias, const float* __restrict__ q,
                           float* __restrict__ out_f32, __bf16* __restrict__ out_bf16,
                           int n_cells) {
  int idx = blockIdx.x * blockDim.x + threadIdx.x;
  if (idx >= n_cells * EMB) return;
  int cell = idx / EMB, m = idx % EMB;
  float c = cnt[cell]; c = c > 1.0f ? c : 1.0f;
  float v = agg[idx] / c + bias[m] + q[idx];
  v = v > 0.0f ? v : (__expf(v) - 1.0f);                 // ELU (alpha=1)
  if (out_f32)  out_f32[idx]  = v;
  if (out_bf16) out_bf16[idx] = (__bf16)v;
}

// ---------------------------------------------------------------------------
extern "C" void kernel_launch(void* const* d_in, const int* in_sizes, int n_in,
                              void* d_out, int out_size, void* d_ws, size_t ws_size,
                              hipStream_t stream) {
  const float* x   = (const float*)d_in[0];
  const int*   ei  = (const int*)d_in[1];
  const float* Wl0 = (const float*)d_in[2];
  const float* bl0 = (const float*)d_in[3];
  const float* Wr0 = (const float*)d_in[4];
  const float* Wl1 = (const float*)d_in[5];
  const float* bl1 = (const float*)d_in[6];
  const float* Wr1 = (const float*)d_in[7];

  const int emb     = in_sizes[3];            // 128
  const int in_dim  = in_sizes[2] / emb;      // 1000
  const int n       = in_sizes[0] / in_dim;   // 50000
  const int n_edges = in_sizes[1] / 2;        // 800000
  const int Kpad0   = ((in_dim + 31) / 32) * 32;   // 1024
  const int Kpad1   = emb;                    // 128 (multiple of 32)

  const int* src = ei;
  const int* dst = ei + n_edges;

  // ---- workspace carve-up (256B aligned) ----
  char* wsp = (char*)d_ws;
  auto carve = [&](size_t bytes) {
    char* p = wsp; wsp += (bytes + 255) & ~(size_t)255; return (void*)p;
  };
  __bf16* xb   = (__bf16*)carve((size_t)n * Kpad0 * 2);      // ~102 MB
  __bf16* Wl0b = (__bf16*)carve((size_t)emb * Kpad0 * 2);
  __bf16* Wr0b = (__bf16*)carve((size_t)emb * Kpad0 * 2);
  __bf16* Wl1b = (__bf16*)carve((size_t)emb * Kpad1 * 2);
  __bf16* Wr1b = (__bf16*)carve((size_t)emb * Kpad1 * 2);
  float*  p    = (float*) carve((size_t)n * emb * 4);
  float*  q    = (float*) carve((size_t)n * emb * 4);
  float*  agg  = (float*) carve((size_t)n * emb * 4);
  float*  cnt  = (float*) carve((size_t)n * 4);
  __bf16* h1b  = (__bf16*)carve((size_t)n * emb * 2);

  const size_t nf = (size_t)n * emb;
  const int    fin_blocks  = (int)((nf + 255) / 256);
  const int    gemm_blocks = (n + 15) / 16;
  const long long sthreads = (long long)n_edges * 32;
  const int    scat_blocks = (int)((sthreads + 255) / 256);

  // 1) repack x -> bf16 [cell][Kpad0]; convert weights (zero-padded K)
  dim3 tb(32, 8), tg((n + 31) / 32, (Kpad0 + 31) / 32);
  k_transpose_bf16<<<tg, tb, 0, stream>>>(x, xb, in_dim, n, Kpad0);
  int wc0 = emb * Kpad0, wc1 = emb * Kpad1;
  k_wconvert<<<(wc0 + 255) / 256, 256, 0, stream>>>(Wl0, Wl0b, emb, in_dim, Kpad0);
  k_wconvert<<<(wc0 + 255) / 256, 256, 0, stream>>>(Wr0, Wr0b, emb, in_dim, Kpad0);
  k_wconvert<<<(wc1 + 255) / 256, 256, 0, stream>>>(Wl1, Wl1b, emb, emb, Kpad1);
  k_wconvert<<<(wc1 + 255) / 256, 256, 0, stream>>>(Wr1, Wr1b, emb, emb, Kpad1);

  // 2) in-degree counts (same graph for both layers -> compute once)
  k_zero <<<(n + 255) / 256, 256, 0, stream>>>(cnt, (size_t)n);
  k_count<<<(n_edges + 255) / 256, 256, 0, stream>>>(dst, cnt, n_edges);

  // 3) layer 0: project first (8x less scatter traffic), then aggregate
  k_gemm_wmma<<<gemm_blocks, 256, 0, stream>>>(Wl0b, xb, p, n, Kpad0);
  k_gemm_wmma<<<gemm_blocks, 256, 0, stream>>>(Wr0b, xb, q, n, Kpad0);
  k_zero    <<<fin_blocks, 256, 0, stream>>>(agg, nf);
  k_scatter <<<scat_blocks, 256, 0, stream>>>(src, dst, p, agg, n_edges);
  k_finalize<<<fin_blocks, 256, 0, stream>>>(agg, cnt, bl0, q, nullptr, h1b, n);

  // 4) layer 1 (K = 128, same kernels)
  k_gemm_wmma<<<gemm_blocks, 256, 0, stream>>>(Wl1b, h1b, p, n, Kpad1);
  k_gemm_wmma<<<gemm_blocks, 256, 0, stream>>>(Wr1b, h1b, q, n, Kpad1);
  k_zero    <<<fin_blocks, 256, 0, stream>>>(agg, nf);
  k_scatter <<<scat_blocks, 256, 0, stream>>>(src, dst, p, agg, n_edges);
  k_finalize<<<fin_blocks, 256, 0, stream>>>(agg, cnt, bl1, q, (float*)d_out, nullptr, n);
}
